// PointWiseGate_63170378990046
// MI455X (gfx1250) — compile-verified, matrix operand order checked by reading, not run on hardware
//
#include <hip/hip_runtime.h>
#include <hip/hip_bf16.h>
#include <math.h>

// ---------------- problem constants (from setup_inputs) ----------------
#define BATCH 4
#define NPTS  16384
#define CI    256
#define CP    256
#define CIN   512     // CI + CP
#define CB    64      // bottleneck
#define IH    64
#define IW    160

// ---------------- WMMA types (gfx1250, wave32) ----------------
typedef __attribute__((ext_vector_type(16))) __bf16 v16bf;
typedef __attribute__((ext_vector_type(8)))  __bf16 v8bf;
typedef __attribute__((ext_vector_type(8)))  float  v8f;

union V16U { v16bf v; v8bf h[2]; };
__device__ __forceinline__ v16bf make16(v8bf lo, v8bf hi) {
    V16U u; u.h[0] = lo; u.h[1] = hi; return u.v;
}
__device__ __forceinline__ v8f wmma_bf16(v16bf a, v16bf b, v8f c) {
    return __builtin_amdgcn_wmma_f32_16x16x32_bf16(false, a, false, b, (short)0, c, false, false);
}

// ---------------- optional CDNA5 async global->LDS path ----------------
#if defined(__has_builtin)
# if __has_builtin(__builtin_amdgcn_global_load_async_to_lds_b128)
#  define HAVE_ASYNC_COPY 1
# endif
# if __has_builtin(__builtin_amdgcn_s_wait_asynccnt)
#  define HAVE_WAIT_ASYNC 1
# endif
#endif

#if defined(HAVE_ASYNC_COPY)
// builtin signature: (v4i addrspace(1)*, v4i addrspace(3)*, imm offset, imm cpol)
typedef __attribute__((ext_vector_type(4))) int i32x4;
typedef __attribute__((address_space(1))) i32x4 gi32x4;
typedef __attribute__((address_space(3))) i32x4 li32x4;
__device__ __forceinline__ void async_cp16(const void* g, void* l) {
    // per-lane 16B async copy: GLOBAL_LOAD_ASYNC_TO_LDS_B128 (ASYNCcnt tracked)
    __builtin_amdgcn_global_load_async_to_lds_b128(
        (gi32x4*)(uintptr_t)g, (li32x4*)(uintptr_t)l, 0, 0);
}
__device__ __forceinline__ void wait_stage() {
# if defined(HAVE_WAIT_ASYNC)
    __builtin_amdgcn_s_wait_asynccnt(0);
# else
    asm volatile("s_wait_asynccnt 0x0" ::: "memory");
# endif
}
#else
__device__ __forceinline__ void wait_stage() {}
#endif

// stage one W K-slice row (64B) for this thread into LDS
__device__ __forceinline__ void stage_row(const __bf16* __restrict__ Wbf,
                                          __bf16* dst, int k0, int tid) {
    const __bf16* src = Wbf + (size_t)tid*CIN + k0;
    __bf16* d = dst + tid*32;
#if defined(HAVE_ASYNC_COPY)
    async_cp16(src,      d);
    async_cp16(src + 8,  d + 8);
    async_cp16(src + 16, d + 16);
    async_cp16(src + 24, d + 24);
#else
    ((v8bf*)d)[0] = ((const v8bf*)src)[0];
    ((v8bf*)d)[1] = ((const v8bf*)src)[1];
    ((v8bf*)d)[2] = ((const v8bf*)src)[2];
    ((v8bf*)d)[3] = ((const v8bf*)src)[3];
#endif
}

// ---------------- workspace layout (bytes) ----------------
static constexpr size_t OFF_WFUS  = 0;                               // 256x512 bf16
static constexpr size_t OFF_SAW   = OFF_WFUS + (size_t)CP*CIN*2;     // 64x256 bf16
static constexpr size_t OFF_CA1W  = OFF_SAW  + (size_t)CB*CP*2;
static constexpr size_t OFF_CA2W  = OFF_CA1W + (size_t)CB*CP*2;      // 256x64 bf16
static constexpr size_t OFF_BNSC  = OFF_CA2W + (size_t)CP*CB*2;
static constexpr size_t OFF_BNSH  = OFF_BNSC + (size_t)CP*4;
static constexpr size_t OFF_CAMAX = OFF_BNSH + (size_t)CP*4;
static constexpr size_t OFF_XT    = (OFF_CAMAX + (size_t)BATCH*CP*4 + 255) & ~(size_t)255;
// X (bf16, B*N*512) and G (f32, B*256*N) are the same 67.1MB; X dead after GEMM -> alias.
static constexpr size_t OFF_Y     = OFF_XT + (size_t)BATCH*NPTS*CIN*2;

// ---------------- K0: fp32 -> bf16 weight convert ----------------
__global__ __launch_bounds__(256) void cvt_bf16(const float* __restrict__ src,
                                                __bf16* __restrict__ dst, int n) {
    int i = blockIdx.x * 256 + threadIdx.x;
    if (i < n) dst[i] = (__bf16)src[i];
}

// ---------------- K1: bilinear grid-sample gather + concat pack ----------------
// Output Xt[b][n][k] bf16, k contiguous (k<256: sampled img feats, k>=256: pts feats).
__global__ __launch_bounds__(256) void gather_pack(const float* __restrict__ pts_img,
                                                   const float* __restrict__ img,
                                                   const float* __restrict__ pts,
                                                   __bf16* __restrict__ Xt) {
    __shared__ int   sx0[32], sy0[32], sb[32], sn[32];
    __shared__ float swx[32], swy[32];
    const int t = threadIdx.x;
    if (t < 32) {
        int pg = blockIdx.x * 32 + t;
        int b = pg / NPTS, n = pg % NPTS;
        float gx = pts_img[(size_t)pg*2 + 0];
        float gy = pts_img[(size_t)pg*2 + 1];
        float x = ((gx + 1.0f) * (float)IW - 1.0f) * 0.5f;   // align_corners=False
        float y = ((gy + 1.0f) * (float)IH - 1.0f) * 0.5f;
        float x0f = floorf(x), y0f = floorf(y);
        sx0[t] = (int)x0f; sy0[t] = (int)y0f;
        swx[t] = x - x0f;  swy[t] = y - y0f;
        sb[t] = b; sn[t] = n;
    }
    __syncthreads();
    const int c = t;   // 256 threads == 256 channels
    for (int p = 0; p < 32; ++p) {
        const int b = sb[p], n = sn[p], x0 = sx0[p], y0 = sy0[p];
        const float wx1 = swx[p], wy1 = swy[p];
        const float wx0 = 1.0f - wx1, wy0 = 1.0f - wy1;
        const int x1 = x0 + 1, y1 = y0 + 1;
        const bool vx0 = (x0 >= 0) & (x0 < IW), vx1 = (x1 >= 0) & (x1 < IW);
        const bool vy0 = (y0 >= 0) & (y0 < IH), vy1 = (y1 >= 0) & (y1 < IH);
        const float* base = img + ((size_t)b*CI + c) * (IH*IW);
        float acc = 0.0f;  // zero padding == skip invalid corners
        if (vx0 & vy0) acc += wx0*wy0 * base[y0*IW + x0];
        if (vx1 & vy0) acc += wx1*wy0 * base[y0*IW + x1];
        if (vx0 & vy1) acc += wx0*wy1 * base[y1*IW + x0];
        if (vx1 & vy1) acc += wx1*wy1 * base[y1*IW + x1];
        __bf16* xc = Xt + ((size_t)b*NPTS + n) * CIN;
        xc[c]      = (__bf16)acc;
        xc[CI + c] = (__bf16)pts[((size_t)b*CP + c)*NPTS + n];
    }
}

// ---------------- K2: fusion GEMM  Y[b] = W(256x512) @ X[b](512xN) ----------------
// 8 waves/block = 4 column-groups x 2 M-halves; wave owns 8 M-tiles x 16 columns.
// (64 acc VGPRs + <=64 A-preload + B regs stays well under 256 VGPRs -> no
//  s_set_vgpr_msb shuffles.)  W K-slices double-buffered in LDS via async
// global->LDS copies; s_wait_asynccnt + barrier close each K-step.
// fusion_b intentionally omitted: a per-channel constant cancels in training-mode BN.
__global__ __launch_bounds__(256) void gemm_fusion(const __bf16* __restrict__ Wbf,
                                                   const __bf16* __restrict__ Xt,
                                                   float* __restrict__ Y) {
    __shared__ __bf16 sW[2][256 * 32];              // 2 x (W[:, k0:k0+32]) = 2 x 16KB
    const int nbn    = NPTS / 64;
    const int b      = blockIdx.x / nbn;
    const int nblk   = blockIdx.x % nbn;
    const int wave   = threadIdx.x >> 5;
    const int lane   = threadIdx.x & 31;
    const int colgrp = wave & 3;                    // 4 column groups of 16
    const int mhalf  = wave >> 2;                   // 0: rows 0..127, 1: rows 128..255
    const int hlf    = lane >> 4;                   // 0/1: K-phase half
    const int l16    = lane & 15;
    const int ncol   = nblk*64 + colgrp*16 + l16;   // this lane's output column
    const __bf16* xcol = Xt + ((size_t)b*NPTS + ncol) * CIN;
    const int kb = hlf * 8;                         // lanes>=16 start at K+8 (ISA layout)

    const v8f vz = {0.f,0.f,0.f,0.f,0.f,0.f,0.f,0.f};
    v8f acc[8];
    #pragma unroll
    for (int i = 0; i < 8; ++i) acc[i] = vz;

    // prologue: stage slice 0
    stage_row(Wbf, sW[0], 0, threadIdx.x);
    wait_stage();
    __syncthreads();

    #pragma unroll 1
    for (int ks = 0; ks < CIN/32; ++ks) {
        const int buf = ks & 1;
        const int k0  = ks * 32;
        if (ks + 1 < CIN/32)                        // stage next slice into other buffer
            stage_row(Wbf, sW[buf ^ 1], k0 + 32, threadIdx.x);
        // B operand: two contiguous b128 loads per lane, reused for 8 WMMAs
        v16bf Bm = make16(*(const v8bf*)(xcol + k0 + kb),
                          *(const v8bf*)(xcol + k0 + kb + 16));
        if (ks + 1 < CIN/32) __builtin_prefetch(xcol + k0 + 32 + kb, 0, 0);
        #pragma unroll
        for (int mt = 0; mt < 8; ++mt) {
            const int m = mhalf*128 + mt*16 + l16;
            v16bf Am = make16(*(const v8bf*)(&sW[buf][m*32 + kb]),
                              *(const v8bf*)(&sW[buf][m*32 + kb + 16]));
            acc[mt] = wmma_bf16(Am, Bm, acc[mt]);
        }
        wait_stage();                               // own async copies complete
        __syncthreads();                            // whole block's slice visible
    }
    #pragma unroll
    for (int mt = 0; mt < 8; ++mt)
        #pragma unroll
        for (int v = 0; v < 8; ++v) {
            const int row = mhalf*128 + mt*16 + v + hlf*8;   // ISA C/D layout
            Y[((size_t)b*CP + row)*NPTS + ncol] = acc[mt][v];
        }
}

// ---------------- K3: training-mode BN stats (per channel over B*N) ----------------
__global__ __launch_bounds__(256) void bn_stats(const float* __restrict__ Y,
                                                const float* __restrict__ gamma,
                                                const float* __restrict__ beta,
                                                float* __restrict__ bnscale,
                                                float* __restrict__ bnshift,
                                                float* __restrict__ camax) {
    const int c = blockIdx.x;
    float s = 0.f, s2 = 0.f;
    for (int b = 0; b < BATCH; ++b) {
        const float* row = Y + ((size_t)b*CP + c)*NPTS;
        for (int n = threadIdx.x; n < NPTS; n += 256) { float v = row[n]; s += v; s2 += v*v; }
    }
    __shared__ float rs[256], rs2[256];
    rs[threadIdx.x] = s; rs2[threadIdx.x] = s2; __syncthreads();
    for (int off = 128; off > 0; off >>= 1) {
        if (threadIdx.x < off) { rs[threadIdx.x] += rs[threadIdx.x+off]; rs2[threadIdx.x] += rs2[threadIdx.x+off]; }
        __syncthreads();
    }
    if (threadIdx.x == 0) {
        const float inv = 1.0f / (float)(BATCH * NPTS);
        float mean = rs[0] * inv;
        float var  = rs2[0] * inv - mean*mean;          // biased variance
        float sc   = gamma[c] * rsqrtf(var + 1e-5f);
        bnscale[c] = sc;
        bnshift[c] = beta[c] - mean*sc;
        for (int b = 0; b < BATCH; ++b) camax[b*CP + c] = -3.0e38f;
    }
}

// ---------------- K4: BN+ReLU + spatial attn + channel-attn GEMMs ----------------
// Block covers 64 columns. All three small GEMMs run on WMMA against LDS tiles.
__global__ __launch_bounds__(256) void attention_fused(
        const float* __restrict__ Y,
        const float* __restrict__ bnscale, const float* __restrict__ bnshift,
        const __bf16* __restrict__ saW,  const float* __restrict__ saB,
        const __bf16* __restrict__ ca1W, const float* __restrict__ ca1B,
        const __bf16* __restrict__ ca2W, const float* __restrict__ ca2B,
        float* __restrict__ G, float* __restrict__ camax) {
    __shared__ __bf16 sFB[64 * 256];   // [n][k] bf16 F (then G) tile, 32KB
    __shared__ __bf16 sH [64 * 64];    // [n][k] bf16 ca1 output, 8KB
    __shared__ float  sSp[64];
    const int nbn = NPTS / 64;
    const int b   = blockIdx.x / nbn;
    const int n0  = (blockIdx.x % nbn) * 64;
    const int t = threadIdx.x;
    const int wave = t >> 5, lane = t & 31, hlf = lane >> 4, l16 = lane & 15;
    const int kb = hlf * 8;
    const int cq = t >> 4;            // 0..15 row group
    const int cj = (t & 15) * 4;      // column offset (4 cols/thread)
    const v8f vz = {0.f,0.f,0.f,0.f,0.f,0.f,0.f,0.f};

    // Step 1: BN + ReLU -> bf16 B-operand tile
    #pragma unroll 4
    for (int rep = 0; rep < 16; ++rep) {
        const int c = rep*16 + cq;
        const float4 yv = *(const float4*)(Y + ((size_t)b*CP + c)*NPTS + n0 + cj);
        const float sc = bnscale[c], sh = bnshift[c];
        sFB[(cj+0)*256 + c] = (__bf16)fmaxf(yv.x*sc + sh, 0.f);
        sFB[(cj+1)*256 + c] = (__bf16)fmaxf(yv.y*sc + sh, 0.f);
        sFB[(cj+2)*256 + c] = (__bf16)fmaxf(yv.z*sc + sh, 0.f);
        sFB[(cj+3)*256 + c] = (__bf16)fmaxf(yv.w*sc + sh, 0.f);
    }
    if (t < 64) sSp[t] = -3.0e38f;
    __syncthreads();

    // Step 2: spatial attention GEMM (64x256 @ 256x64), column-max + sigmoid
    #pragma unroll
    for (int j = 0; j < 2; ++j) {
        const int id = wave*2 + j;
        const int mt = id >> 2, nt = id & 3;
        const __bf16* arow = saW + (size_t)(mt*16 + l16)*CP;
        const __bf16* bcol = sFB + (nt*16 + l16)*256;
        v8f acc = vz;
        #pragma unroll
        for (int k0 = 0; k0 < CP; k0 += 32) {
            v16bf Am = make16(*(const v8bf*)(arow + k0 + kb), *(const v8bf*)(arow + k0 + kb + 16));
            v16bf Bm = make16(*(const v8bf*)(bcol + k0 + kb), *(const v8bf*)(bcol + k0 + kb + 16));
            acc = wmma_bf16(Am, Bm, acc);
        }
        float cm = -3.0e38f;
        #pragma unroll
        for (int v = 0; v < 8; ++v)
            cm = fmaxf(cm, acc[v] + saB[mt*16 + v + hlf*8]);
        cm = fmaxf(cm, __shfl_xor(cm, 16, 32));        // merge row halves (same col)
        if (hlf == 0) atomicMax(&sSp[nt*16 + l16], cm);
    }
    __syncthreads();
    if (t < 64) sSp[t] = 1.0f / (1.0f + expf(-sSp[t]));
    __syncthreads();

    // Step 3: G = spatial * F ; store fp32 G, refresh bf16 tile in place
    #pragma unroll 4
    for (int rep = 0; rep < 16; ++rep) {
        const int c = rep*16 + cq;
        const float g0 = (float)sFB[(cj+0)*256 + c] * sSp[cj+0];
        const float g1 = (float)sFB[(cj+1)*256 + c] * sSp[cj+1];
        const float g2 = (float)sFB[(cj+2)*256 + c] * sSp[cj+2];
        const float g3 = (float)sFB[(cj+3)*256 + c] * sSp[cj+3];
        *(float4*)(G + ((size_t)b*CP + c)*NPTS + n0 + cj) = make_float4(g0,g1,g2,g3);
        sFB[(cj+0)*256 + c] = (__bf16)g0;
        sFB[(cj+1)*256 + c] = (__bf16)g1;
        sFB[(cj+2)*256 + c] = (__bf16)g2;
        sFB[(cj+3)*256 + c] = (__bf16)g3;
    }
    __syncthreads();

    // Step 4: ca1 GEMM (64x256 @ 256x64) + bias + ReLU -> sH[n][k] bf16
    #pragma unroll
    for (int j = 0; j < 2; ++j) {
        const int id = wave*2 + j;
        const int mt = id >> 2, nt = id & 3;
        const __bf16* arow = ca1W + (size_t)(mt*16 + l16)*CP;
        const __bf16* bcol = sFB + (nt*16 + l16)*256;
        v8f acc = vz;
        #pragma unroll
        for (int k0 = 0; k0 < CP; k0 += 32) {
            v16bf Am = make16(*(const v8bf*)(arow + k0 + kb), *(const v8bf*)(arow + k0 + kb + 16));
            v16bf Bm = make16(*(const v8bf*)(bcol + k0 + kb), *(const v8bf*)(bcol + k0 + kb + 16));
            acc = wmma_bf16(Am, Bm, acc);
        }
        #pragma unroll
        for (int v = 0; v < 8; ++v) {
            const int row = mt*16 + v + hlf*8;
            sH[(nt*16 + l16)*64 + row] = (__bf16)fmaxf(acc[v] + ca1B[row], 0.f);
        }
    }
    __syncthreads();

    // Step 5: ca2 GEMM (256x64 @ 64x64) + bias, per-channel max over cols -> global atomic
    #pragma unroll
    for (int j = 0; j < 8; ++j) {
        const int id = wave*8 + j;
        const int mt = id >> 2, nt = id & 3;
        const __bf16* arow = ca2W + (size_t)(mt*16 + l16)*CB;
        const __bf16* bcol = sH  + (nt*16 + l16)*64;
        v8f acc = vz;
        #pragma unroll
        for (int k0 = 0; k0 < CB; k0 += 32) {
            v16bf Am = make16(*(const v8bf*)(arow + k0 + kb), *(const v8bf*)(arow + k0 + kb + 16));
            v16bf Bm = make16(*(const v8bf*)(bcol + k0 + kb), *(const v8bf*)(bcol + k0 + kb + 16));
            acc = wmma_bf16(Am, Bm, acc);
        }
        #pragma unroll
        for (int v = 0; v < 8; ++v) {
            const int row = mt*16 + v + hlf*8;
            float val = acc[v] + ca2B[row];
            #pragma unroll
            for (int off = 1; off < 16; off <<= 1)
                val = fmaxf(val, __shfl_xor(val, off, 32));  // max over 16 cols
            if (l16 == 0) atomicMax(&camax[b*CP + row], val);
        }
    }
}

// ---------------- K5: out = sigmoid(channel_max) * G ----------------
__global__ __launch_bounds__(256) void final_gate(const float* __restrict__ G,
                                                  const float* __restrict__ camax,
                                                  float* __restrict__ out) {
    const size_t i4 = (size_t)blockIdx.x * 256 + threadIdx.x;
    const size_t flat = i4 * 4;
    const int bc = (int)(flat / NPTS);               // b*CP + c (N%4==0)
    const float gate = 1.0f / (1.0f + expf(-camax[bc]));
    const float4 g = *(const float4*)(G + flat);
    *(float4*)(out + flat) = make_float4(g.x*gate, g.y*gate, g.z*gate, g.w*gate);
}

// ---------------- host launcher ----------------
extern "C" void kernel_launch(void* const* d_in, const int* in_sizes, int n_in,
                              void* d_out, int out_size, void* d_ws, size_t ws_size,
                              hipStream_t stream) {
    const float* pts_img  = (const float*)d_in[0];
    const float* img      = (const float*)d_in[1];
    const float* pts      = (const float*)d_in[2];
    const float* fusion_w = (const float*)d_in[3];
    // d_in[4] fusion_b: mathematically cancels under training-mode BatchNorm.
    const float* bn_gamma = (const float*)d_in[5];
    const float* bn_beta  = (const float*)d_in[6];
    const float* ca1_w    = (const float*)d_in[7];
    const float* ca1_b    = (const float*)d_in[8];
    const float* ca2_w    = (const float*)d_in[9];
    const float* ca2_b    = (const float*)d_in[10];
    const float* sa_w     = (const float*)d_in[11];
    const float* sa_b     = (const float*)d_in[12];

    char* ws = (char*)d_ws;
    __bf16* wfus   = (__bf16*)(ws + OFF_WFUS);
    __bf16* saw    = (__bf16*)(ws + OFF_SAW);
    __bf16* ca1w   = (__bf16*)(ws + OFF_CA1W);
    __bf16* ca2w   = (__bf16*)(ws + OFF_CA2W);
    float*  bnsc   = (float*) (ws + OFF_BNSC);
    float*  bnsh   = (float*) (ws + OFF_BNSH);
    float*  camax  = (float*) (ws + OFF_CAMAX);
    __bf16* xt     = (__bf16*)(ws + OFF_XT);
    float*  gbuf   = (float*) (ws + OFF_XT);   // aliases xt: X dead after GEMM
    float*  ybuf   = (float*) (ws + OFF_Y);

    cvt_bf16<<<(CP*CIN + 255)/256, 256, 0, stream>>>(fusion_w, wfus, CP*CIN);
    cvt_bf16<<<(CB*CP  + 255)/256, 256, 0, stream>>>(sa_w,  saw,  CB*CP);
    cvt_bf16<<<(CB*CP  + 255)/256, 256, 0, stream>>>(ca1_w, ca1w, CB*CP);
    cvt_bf16<<<(CP*CB  + 255)/256, 256, 0, stream>>>(ca2_w, ca2w, CP*CB);

    gather_pack<<<BATCH*NPTS/32, 256, 0, stream>>>(pts_img, img, pts, xt);
    gemm_fusion<<<BATCH*(NPTS/64), 256, 0, stream>>>(wfus, xt, ybuf);
    bn_stats<<<CP, 256, 0, stream>>>(ybuf, bn_gamma, bn_beta, bnsc, bnsh, camax);
    attention_fused<<<BATCH*(NPTS/64), 256, 0, stream>>>(ybuf, bnsc, bnsh,
                                                         saw, sa_b, ca1w, ca1_b,
                                                         ca2w, ca2_b, gbuf, camax);
    final_gate<<<(BATCH*CP*NPTS/4)/256, 256, 0, stream>>>(gbuf, camax, (float*)d_out);
}
